// GAU_66503273612026
// MI455X (gfx1250) — matrix operand order repaired and checked by reading, not compiled, not run on hardware
//
#include <hip/hip_runtime.h>
#include <hip/hip_bf16.h>
#include <stdint.h>

typedef __bf16 bf16_t;
typedef __attribute__((ext_vector_type(8)))  bf16_t bf16x8;
typedef __attribute__((ext_vector_type(16))) bf16_t bf16x16;
typedef __attribute__((ext_vector_type(8)))  float  v8f;

// ---------------------------------------------------------------- helpers ----

static __device__ __forceinline__ unsigned short f2bf(float f) {
  unsigned u = __float_as_uint(f);
  unsigned r = (u + 0x7FFFu + ((u >> 16) & 1u)) >> 16;  // RNE
  return (unsigned short)r;
}

static __device__ __forceinline__ float silu_f(float x) {
  return x / (1.0f + __expf(-x));
}

static __device__ __forceinline__ v8f wmma_bf16(bf16x16 a, bf16x16 b, v8f c) {
  return __builtin_amdgcn_wmma_f32_16x16x32_bf16(false, a, false, b, (short)0, c,
                                                 false, false);
}

// A-fragment (16x32 bf16, MxK) from a row-major matrix.
// Lane L: M = L&15, half = L>>4. Half 0 takes K in {kt..kt+15},
// half 1 takes {kt+16..kt+31} -> one contiguous 32B chunk (matches B layout).
static __device__ __forceinline__ bf16x16 load_a_frag(const bf16_t* __restrict__ A,
                                                      int row, int ldk, int kt, int half) {
  const bf16x8* p = reinterpret_cast<const bf16x8*>(A + (size_t)row * ldk + kt + half * 16);
  return __builtin_shufflevector(p[0], p[1],
                                 0, 1, 2, 3, 4, 5, 6, 7, 8, 9, 10, 11, 12, 13, 14, 15);
}

// B-fragment (32x16 bf16, KxN) from the pre-swizzled weight layout:
// tile (ktile,ntile) stores 512 bf16; each lane owns 32 contiguous bytes.
static __device__ __forceinline__ bf16x16 load_b_sw(const bf16_t* __restrict__ Bsw,
                                                    int ktile, int ntile, int Ntiles,
                                                    int lane) {
  const bf16x8* p = reinterpret_cast<const bf16x8*>(
      Bsw + ((size_t)(ktile * Ntiles + ntile)) * 512 + lane * 16);
  return __builtin_shufflevector(p[0], p[1],
                                 0, 1, 2, 3, 4, 5, 6, 7, 8, 9, 10, 11, 12, 13, 14, 15);
}

// B-fragment for Q@K^T: B[k, j] = K[j, k]; lane's (k,k+1) pairs are contiguous in row j.
// Lane L: column j = base + (L&15); half selects K offset 0 / 16.
static __device__ __forceinline__ bf16x16 load_b_kt(const bf16_t* __restrict__ Kmat,
                                                    int jrow, int ldk, int kt, int half) {
  const bf16x8* p = reinterpret_cast<const bf16x8*>(Kmat + (size_t)jrow * ldk + kt + half * 16);
  return __builtin_shufflevector(p[0], p[1],
                                 0, 1, 2, 3, 4, 5, 6, 7, 8, 9, 10, 11, 12, 13, 14, 15);
}

// ---------------------------------------------------------- prep kernels ----

__global__ __launch_bounds__(256) void k_convert_bf16(const float* __restrict__ src,
                                                      uint4* __restrict__ dst, int n8) {
  int i = blockIdx.x * 256 + threadIdx.x;
  if (i >= n8) return;
  const float4* p = reinterpret_cast<const float4*>(src) + (size_t)i * 2;
  float4 a = p[0], b = p[1];
  uint4 o;
  o.x = (unsigned)f2bf(a.x) | ((unsigned)f2bf(a.y) << 16);
  o.y = (unsigned)f2bf(a.z) | ((unsigned)f2bf(a.w) << 16);
  o.z = (unsigned)f2bf(b.x) | ((unsigned)f2bf(b.y) << 16);
  o.w = (unsigned)f2bf(b.z) | ((unsigned)f2bf(b.w) << 16);
  dst[i] = o;
}

// W (K x N, f32 row-major) -> bf16 B-fragment-swizzled layout.
// dword id -> (ktile, ntile, lane, r); stores pair (W[k0,n], W[k0+1,n]).
__global__ __launch_bounds__(256) void k_swizzle_w(const float* __restrict__ W,
                                                   unsigned* __restrict__ Wsw, int N) {
  int id = blockIdx.x * 256 + threadIdx.x;
  int r = id & 7;
  int lane = (id >> 3) & 31;
  int tile = id >> 8;
  int Ntiles = N >> 4;
  int ntile = tile % Ntiles;
  int ktile = tile / Ntiles;
  int k0 = ktile * 32 + (lane >> 4) * 16 + r * 2;
  int n = ntile * 16 + (lane & 15);
  float w0 = W[(size_t)k0 * N + n];
  float w1 = W[((size_t)k0 + 1) * N + n];
  Wsw[id] = (unsigned)f2bf(w0) | ((unsigned)f2bf(w1) << 16);
}

// ------------------------------------------------- GEMM1: vg = silu(v)*silu(g)

__global__ __launch_bounds__(256) void k_gemm_hidden(const bf16_t* __restrict__ xb,
                                                     const bf16_t* __restrict__ whs,
                                                     const float* __restrict__ bh,
                                                     unsigned short* __restrict__ vg) {
  const int K = 1024, Ntiles = 128;  // W_hidden has N = 2048
  int lane = threadIdx.x & 31, wave = threadIdx.x >> 5;
  int l15 = lane & 15, half = lane >> 4;
  int m0 = blockIdx.y * 128 + (wave >> 1) * 32;
  int n0 = blockIdx.x * 64 + (wave & 1) * 32;  // v column; gate at n0+1024
  int nt0 = n0 >> 4;

  v8f av[2][2] = {};
  v8f ag[2][2] = {};
  for (int kt = 0; kt < K; kt += 32) {
    int ktile = kt >> 5;
    bf16x16 a0 = load_a_frag(xb, m0 + l15, K, kt, half);
    bf16x16 a1 = load_a_frag(xb, m0 + 16 + l15, K, kt, half);
#pragma unroll
    for (int ni = 0; ni < 2; ++ni) {
      bf16x16 bv = load_b_sw(whs, ktile, nt0 + ni, Ntiles, lane);
      bf16x16 bg = load_b_sw(whs, ktile, nt0 + 64 + ni, Ntiles, lane);
      av[0][ni] = wmma_bf16(a0, bv, av[0][ni]);
      av[1][ni] = wmma_bf16(a1, bv, av[1][ni]);
      ag[0][ni] = wmma_bf16(a0, bg, ag[0][ni]);
      ag[1][ni] = wmma_bf16(a1, bg, ag[1][ni]);
    }
  }
#pragma unroll
  for (int mi = 0; mi < 2; ++mi)
#pragma unroll
    for (int ni = 0; ni < 2; ++ni) {
      int n = n0 + ni * 16 + l15;
      float biasv = bh[n], biasg = bh[n + 1024];
#pragma unroll
      for (int g = 0; g < 8; ++g) {
        int m = m0 + mi * 16 + half * 8 + g;
        float v = silu_f(av[mi][ni][g] + biasv);
        float gt = silu_f(ag[mi][ni][g] + biasg);
        vg[(size_t)m * 1024 + n] = f2bf(v * gt);
      }
    }
}

// ------------------------------------------------- GEMM2: q,k from z ---------

__global__ __launch_bounds__(256) void k_gemm_qk(const bf16_t* __restrict__ xb,
                                                 const bf16_t* __restrict__ wqs,
                                                 const float* __restrict__ bq,
                                                 const float* __restrict__ gamma,
                                                 const float* __restrict__ beta,
                                                 unsigned short* __restrict__ qb,
                                                 unsigned short* __restrict__ kb) {
  const int K = 1024, Ntiles = 64;
  int lane = threadIdx.x & 31, wave = threadIdx.x >> 5;
  int l15 = lane & 15, half = lane >> 4;
  int m0 = blockIdx.y * 128 + (wave >> 1) * 32;
  int n0 = blockIdx.x * 64 + (wave & 1) * 32;
  int nt0 = n0 >> 4;

  v8f acc[2][2] = {};
  for (int kt = 0; kt < K; kt += 32) {
    int ktile = kt >> 5;
    bf16x16 a0 = load_a_frag(xb, m0 + l15, K, kt, half);
    bf16x16 a1 = load_a_frag(xb, m0 + 16 + l15, K, kt, half);
#pragma unroll
    for (int ni = 0; ni < 2; ++ni) {
      bf16x16 bfr = load_b_sw(wqs, ktile, nt0 + ni, Ntiles, lane);
      acc[0][ni] = wmma_bf16(a0, bfr, acc[0][ni]);
      acc[1][ni] = wmma_bf16(a1, bfr, acc[1][ni]);
    }
  }
#pragma unroll
  for (int mi = 0; mi < 2; ++mi)
#pragma unroll
    for (int ni = 0; ni < 2; ++ni) {
      int n = n0 + ni * 16 + l15;
      float bias = bq[n];
      float g0 = gamma[n], g1 = gamma[n + 1024];
      float be0 = beta[n], be1 = beta[n + 1024];
#pragma unroll
      for (int g = 0; g < 8; ++g) {
        int m = m0 + mi * 16 + half * 8 + g;
        float z = silu_f(acc[mi][ni][g] + bias);
        qb[(size_t)m * 1024 + n] = f2bf(z * g0 + be0);
        kb[(size_t)m * 1024 + n] = f2bf(z * g1 + be1);
      }
    }
}

// --------------------------------- s_ii = (q_i . k_i)/32 --------------------

__global__ __launch_bounds__(256) void k_diag_init(const bf16_t* __restrict__ qb,
                                                   const bf16_t* __restrict__ kb,
                                                   float* __restrict__ sdiag) {
  int wave = threadIdx.x >> 5, lane = threadIdx.x & 31;
  int m = blockIdx.x * 8 + wave;
  const bf16x8* qp = reinterpret_cast<const bf16x8*>(qb + (size_t)m * 1024 + lane * 32);
  const bf16x8* kp = reinterpret_cast<const bf16x8*>(kb + (size_t)m * 1024 + lane * 32);
  float s = 0.f;
#pragma unroll
  for (int c = 0; c < 4; ++c) {
    bf16x8 qv = qp[c], kv = kp[c];
#pragma unroll
    for (int e = 0; e < 8; ++e) s += (float)qv[e] * (float)kv[e];
  }
  s += __shfl_xor(s, 1);
  s += __shfl_xor(s, 2);
  s += __shfl_xor(s, 4);
  s += __shfl_xor(s, 8);
  s += __shfl_xor(s, 16);
  if (lane == 0) sdiag[m] = s * 0.03125f;
}

// ---- attention: dsum_part[slice][i] = sum_{j in slice} exp(s_ij - s_ii) ----
// Deterministic: each (row, slice) pair is owned by exactly one wave; no atomics.

__global__ __launch_bounds__(256) void k_attn(const bf16_t* __restrict__ qb,
                                              const bf16_t* __restrict__ kb,
                                              const float* __restrict__ sdiag,
                                              float* __restrict__ dsum_part) {
  const int K = 1024;
  int lane = threadIdx.x & 31, wave = threadIdx.x >> 5;
  int l15 = lane & 15, half = lane >> 4;
  int batch = blockIdx.y;
  int m0 = batch * 2048 + blockIdx.x * 256 + wave * 32;
  int jbase = blockIdx.z * 512;  // j-slice within the batch
  float* dpart = dsum_part + (size_t)blockIdx.z * 16384;

  float sd[2][8], rs[2][8];
#pragma unroll
  for (int mi = 0; mi < 2; ++mi)
#pragma unroll
    for (int g = 0; g < 8; ++g) {
      sd[mi][g] = sdiag[m0 + mi * 16 + half * 8 + g];
      rs[mi][g] = 0.f;
    }

  for (int j = jbase; j < jbase + 512; j += 32) {
    v8f acc[2][2] = {};
    for (int kt = 0; kt < K; kt += 32) {
      bf16x16 a0 = load_a_frag(qb, m0 + l15, K, kt, half);
      bf16x16 a1 = load_a_frag(qb, m0 + 16 + l15, K, kt, half);
#pragma unroll
      for (int ni = 0; ni < 2; ++ni) {
        int jrow = batch * 2048 + j + ni * 16 + l15;
        bf16x16 bfr = load_b_kt(kb, jrow, K, kt, half);
        acc[0][ni] = wmma_bf16(a0, bfr, acc[0][ni]);
        acc[1][ni] = wmma_bf16(a1, bfr, acc[1][ni]);
      }
    }
#pragma unroll
    for (int mi = 0; mi < 2; ++mi)
#pragma unroll
      for (int ni = 0; ni < 2; ++ni)
#pragma unroll
        for (int g = 0; g < 8; ++g)
          rs[mi][g] += __expf(acc[mi][ni][g] * 0.03125f - sd[mi][g]);
  }
#pragma unroll
  for (int mi = 0; mi < 2; ++mi)
#pragma unroll
    for (int g = 0; g < 8; ++g) {
      float v = rs[mi][g];
      v += __shfl_xor(v, 1);
      v += __shfl_xor(v, 2);
      v += __shfl_xor(v, 4);
      v += __shfl_xor(v, 8);
      if (l15 == 0) dpart[m0 + mi * 16 + half * 8 + g] = v;
    }
}

// --------------- out GEMM: out[b,0,n,t] = vg@Wout * (1/dsum) + b_out ---------

__global__ __launch_bounds__(256) void k_gemm_out(const bf16_t* __restrict__ vg,
                                                  const bf16_t* __restrict__ wos,
                                                  const float* __restrict__ bo,
                                                  const float* __restrict__ dsum_part,
                                                  float* __restrict__ out) {
  const int K = 1024, Ntiles = 64;
  int lane = threadIdx.x & 31, wave = threadIdx.x >> 5;
  int l15 = lane & 15, half = lane >> 4;
  int m0 = blockIdx.y * 128 + (wave >> 1) * 32;
  int n0 = blockIdx.x * 64 + (wave & 1) * 32;
  int nt0 = n0 >> 4;

  v8f acc[2][2] = {};
  for (int kt = 0; kt < K; kt += 32) {
    int ktile = kt >> 5;
    bf16x16 a0 = load_a_frag(vg, m0 + l15, K, kt, half);
    bf16x16 a1 = load_a_frag(vg, m0 + 16 + l15, K, kt, half);
#pragma unroll
    for (int ni = 0; ni < 2; ++ni) {
      bf16x16 bfr = load_b_sw(wos, ktile, nt0 + ni, Ntiles, lane);
      acc[0][ni] = wmma_bf16(a0, bfr, acc[0][ni]);
      acc[1][ni] = wmma_bf16(a1, bfr, acc[1][ni]);
    }
  }
#pragma unroll
  for (int mi = 0; mi < 2; ++mi) {
    int mrow = m0 + mi * 16 + half * 8;  // 8 consecutive rows (t values)
    int b = mrow >> 11;
    int t0 = mrow & 2047;
    // Sum the 4 j-slice partial denominators in a fixed order (deterministic).
    float4 d0 = make_float4(0.f, 0.f, 0.f, 0.f);
    float4 d1 = make_float4(0.f, 0.f, 0.f, 0.f);
#pragma unroll
    for (int s = 0; s < 4; ++s) {
      const float4* dp = reinterpret_cast<const float4*>(dsum_part + (size_t)s * 16384 + mrow);
      float4 p0 = dp[0], p1 = dp[1];
      d0.x += p0.x; d0.y += p0.y; d0.z += p0.z; d0.w += p0.w;
      d1.x += p1.x; d1.y += p1.y; d1.z += p1.z; d1.w += p1.w;
    }
    float i0 = 1.f / d0.x, i1 = 1.f / d0.y, i2 = 1.f / d0.z, i3 = 1.f / d0.w;
    float i4 = 1.f / d1.x, i5 = 1.f / d1.y, i6 = 1.f / d1.z, i7 = 1.f / d1.w;
#pragma unroll
    for (int ni = 0; ni < 2; ++ni) {
      int n = n0 + ni * 16 + l15;
      float bias = bo[n];
      float* dst = out + ((size_t)(b * 1024 + n)) * 2048 + t0;
      float4 o0 = make_float4(acc[mi][ni][0] * i0 + bias, acc[mi][ni][1] * i1 + bias,
                              acc[mi][ni][2] * i2 + bias, acc[mi][ni][3] * i3 + bias);
      float4 o1 = make_float4(acc[mi][ni][4] * i4 + bias, acc[mi][ni][5] * i5 + bias,
                              acc[mi][ni][6] * i6 + bias, acc[mi][ni][7] * i7 + bias);
      *reinterpret_cast<float4*>(dst) = o0;
      *reinterpret_cast<float4*>(dst + 4) = o1;
    }
  }
}

// ---------------------------------------------------------------- launch ----

extern "C" void kernel_launch(void* const* d_in, const int* in_sizes, int n_in,
                              void* d_out, int out_size, void* d_ws, size_t ws_size,
                              hipStream_t stream) {
  const float* x     = (const float*)d_in[0];
  const float* Wh    = (const float*)d_in[1];
  const float* bh    = (const float*)d_in[2];
  const float* Wqk   = (const float*)d_in[3];
  const float* bqk   = (const float*)d_in[4];
  const float* gamma = (const float*)d_in[5];
  const float* beta  = (const float*)d_in[6];
  const float* Wo    = (const float*)d_in[7];
  const float* bo    = (const float*)d_in[8];
  float* out = (float*)d_out;

  const size_t M = 16384;  // B*T
  char* ws = (char*)d_ws;
  size_t off = 0;
  auto carve = [&](size_t bytes) -> void* {
    void* p = ws + off;
    off += (bytes + 255) & ~(size_t)255;
    return p;
  };
  bf16_t* xb    = (bf16_t*)carve(M * 1024 * 2);       // x in bf16
  bf16_t* whs   = (bf16_t*)carve(1024 * 2048 * 2);    // W_hidden swizzled
  bf16_t* wqs   = (bf16_t*)carve(1024 * 1024 * 2);    // W_qk swizzled
  bf16_t* wos   = (bf16_t*)carve(1024 * 1024 * 2);    // W_out swizzled
  bf16_t* vg    = (bf16_t*)carve(M * 1024 * 2);       // silu(v)*silu(gate)
  bf16_t* qb    = (bf16_t*)carve(M * 1024 * 2);       // q bf16
  bf16_t* kb    = (bf16_t*)carve(M * 1024 * 2);       // k bf16
  float*  sdiag = (float*)carve(M * 4);               // s_ii
  float*  dsum  = (float*)carve(4 * M * 4);           // per-j-slice denominators

  k_convert_bf16<<<(M * 1024 / 8) / 256, 256, 0, stream>>>(x, (uint4*)xb,
                                                           (int)(M * 1024 / 8));
  k_swizzle_w<<<(1024 * 2048 / 2) / 256, 256, 0, stream>>>(Wh, (unsigned*)whs, 2048);
  k_swizzle_w<<<(1024 * 1024 / 2) / 256, 256, 0, stream>>>(Wqk, (unsigned*)wqs, 1024);
  k_swizzle_w<<<(1024 * 1024 / 2) / 256, 256, 0, stream>>>(Wo, (unsigned*)wos, 1024);

  k_gemm_hidden<<<dim3(16, 128), 256, 0, stream>>>(xb, whs, bh, (unsigned short*)vg);
  k_gemm_qk<<<dim3(16, 128), 256, 0, stream>>>(xb, wqs, bqk, gamma, beta,
                                               (unsigned short*)qb, (unsigned short*)kb);
  k_diag_init<<<2048, 256, 0, stream>>>(qb, kb, sdiag);
  k_attn<<<dim3(8, 8, 4), 256, 0, stream>>>(qb, kb, sdiag, dsum);
  k_gemm_out<<<dim3(16, 128), 256, 0, stream>>>(vg, wos, bo, dsum, out);
}
